// FullyConnectedTensorProductConv_61684320305858
// MI455X (gfx1250) — compile-verified, matrix operand order checked by reading, not run on hardware
//
#include <hip/hip_runtime.h>
#include <hip/hip_bf16.h>
#include <math.h>

typedef __attribute__((ext_vector_type(16))) __bf16 v16bf;
typedef __attribute__((ext_vector_type(8)))  __bf16 v8bf;
typedef __attribute__((ext_vector_type(8)))  float  v8f;
typedef __attribute__((ext_vector_type(4)))  float  v4f;
typedef __attribute__((ext_vector_type(2)))  float  v2f;

#define NW 4                    // waves per block (128 threads)
#define MUL 16
#define NPATH 5

// Branchless tanh-approx gelu: tanh(t) = 1 - 2/(exp(2t)+1).
// exp overflow -> inf -> rcp -> 0 -> tanh=1 ; exp underflow -> 0 -> tanh=-1.
__device__ __forceinline__ float gelu_tanh(float x) {
    const float k0 = 0.7978845608028654f, k1 = 0.044715f;
    float t = k0 * (x + k1 * x * x * x);
    float e = __expf(2.0f * t);                       // v_exp_f32
    float th = 1.0f - 2.0f * __builtin_amdgcn_rcpf(e + 1.0f);  // v_rcp_f32
    return 0.5f * x * (1.0f + th);
}

// Load one 16x32 bf16 WMMA operand fragment from a row-major [16 x 64] bf16
// matrix (row stride 64). row = lane&15 ; hi = lane>>4 selects K halves.
// Layout per CDNA5 ISA 7.12.2 (16-bit A-matrix 16x32).
__device__ __forceinline__ v16bf ld_frag(const __bf16* base, int row, int kb, int hi) {
    const __bf16* p = base + row * 64 + kb + hi * 8;
    v8bf lo = *(const v8bf*)p;
    v8bf hv = *(const v8bf*)(p + 16);
    v16bf fr;
#pragma unroll
    for (int j = 0; j < 8; ++j) { fr[j] = lo[j]; fr[8 + j] = hv[j]; }
    return fr;
}

__device__ __forceinline__ v8f wmma_bf16(v16bf a, v16bf b, v8f c) {
    return __builtin_amdgcn_wmma_f32_16x16x32_bf16(false, a, false, b, (short)0, c, false, false);
}

// ---------------------------------------------------------------------------
// Kernel 0: zero a float region
__global__ void zero_kernel(float* p, size_t n) {
    size_t i = (size_t)blockIdx.x * blockDim.x + threadIdx.x;
    size_t stride = (size_t)gridDim.x * blockDim.x;
    for (; i < n; i += stride) p[i] = 0.0f;
}

// Kernel 1: W1 [64 x 1280] f32 -> W1T [1280 x 64] bf16
__global__ void w1t_kernel(const float* __restrict__ W1, __bf16* __restrict__ W1T) {
    int idx = blockIdx.x * blockDim.x + threadIdx.x;
    if (idx < 64 * 1280) {
        int c = idx / 1280, col = idx - c * 1280;
        W1T[col * 64 + c] = (__bf16)W1[idx];
    }
}

// ---------------------------------------------------------------------------
// Kernel 2: fused edge MLP (WMMA) + tensor product + scatter-add
__global__ __launch_bounds__(128) void tp_edge_kernel(
    const float* __restrict__ src_feat,   // [n_src, 64]
    const float* __restrict__ edge_sh,    // [n_edge, 4]
    const float* __restrict__ es,         // [n_edge, 64]
    const float* __restrict__ W0,         // [64, 64]
    const float* __restrict__ b0,         // [64]
    const float* __restrict__ b1,         // [1280]
    const __bf16* __restrict__ W1T,       // [1280, 64] bf16 (col-major image)
    const int*  __restrict__ src,
    const int*  __restrict__ dst,
    float* __restrict__ summed,           // [n_dst, 64]
    float* __restrict__ cnt,              // [n_dst]
    int n_edge)
{
    const float A0C = 0.17677669529663687f;   // 1/sqrt(2*16)
    const float A1C = 0.14433756729740643f;   // 1/sqrt(3*16)
    const float INV_SQRT3 = 0.5773502691896258f;
    const float INV_SQRT2 = 0.7071067811865476f;

    __shared__ __bf16 sW0T[64 * 64];                       // W0^T (bf16)
    __shared__ float  sb0[64];
    __shared__ float  sb1[1280];
    __shared__ __align__(16) __bf16 sH[NW][16 * 64];       // gelu(h) tile
    __shared__ __align__(16) float  sXT[NW][16 * 16 * 2];  // {x0,t2}[e][u]
    __shared__ __align__(16) float  sX1[NW][16 * 16 * 4];  // x1[e][u][k] pad4
    __shared__ __align__(16) float  sSH[NW][16 * 4];       // s0,s1x,s1y,s1z
    __shared__ int    sDST[NW][16];

    int tid = threadIdx.x;
    // preamble: stage W0^T (bf16), b0, b1
    for (int idx = tid; idx < 64 * 64; idx += 128) {
        int cin = idx >> 6, cout = idx & 63;
        sW0T[cout * 64 + cin] = (__bf16)W0[idx];
    }
    if (tid < 64) sb0[tid] = b0[tid];
    for (int idx = tid; idx < 1280; idx += 128) sb1[idx] = b1[idx];
    __syncthreads();

    int wave = tid >> 5;
    int lane = tid & 31;
    int r = lane & 15;          // row/col within tile
    int hi = lane >> 4;         // K-half selector

    int gwave = blockIdx.x * NW + wave;
    int nTiles = (n_edge + 15) >> 4;
    int wstride = gridDim.x * NW;

    for (int tile = gwave; tile < nTiles; tile += wstride) {
        int tb = tile << 4;
        // ---------------- prep: gather per-edge vectors into LDS ----------
        int e = tb + r;
        bool ev = (e < n_edge);
        int ec = ev ? e : (n_edge - 1);
        if (hi == 0) {
            const float* shp = edge_sh + (size_t)ec * 4;
            sSH[wave][r * 4 + 0] = shp[0];
            sSH[wave][r * 4 + 1] = shp[1];
            sSH[wave][r * 4 + 2] = shp[2];
            sSH[wave][r * 4 + 3] = shp[3];
            int d = dst[ec];
            sDST[wave][r] = d;
            if (ev) unsafeAtomicAdd(&cnt[d], 1.0f);
        }
        {
            int sidx = src[ec];
            const float* f = src_feat + (size_t)sidx * 64 + hi * 32;
#pragma unroll
            for (int q8 = 0; q8 < 8; ++q8) {
                v4f v = *(const v4f*)(f + q8 * 4);
#pragma unroll
                for (int j = 0; j < 4; ++j) {
                    int fidx = hi * 32 + q8 * 4 + j;
                    float val = v[j];
                    if (fidx < 16) {
                        sXT[wave][(r * 16 + fidx) * 2] = val;       // x0
                    } else {
                        int m = fidx - 16;
                        int u = m / 3, k = m - 3 * u;
                        sX1[wave][(r * 16 + u) * 4 + k] = val;      // x1
                    }
                }
            }
        }
        asm volatile("s_wait_dscnt 0" ::: "memory");
        {
            float sx = sSH[wave][r * 4 + 1];
            float sy = sSH[wave][r * 4 + 2];
            float sz = sSH[wave][r * 4 + 3];
#pragma unroll
            for (int uu = 0; uu < 8; ++uu) {
                int u = hi * 8 + uu;
                const float* xp = &sX1[wave][(r * 16 + u) * 4];
                sXT[wave][(r * 16 + u) * 2 + 1] =
                    INV_SQRT3 * (xp[0] * sx + xp[1] * sy + xp[2] * sz);   // t2
            }
        }

        // ---------------- GEMM1: h = gelu(es @ W0 + b0), 16x64 tile -------
        v16bf A0, A1;
        {
            const float* ep = es + (size_t)ec * 64;
#pragma unroll
            for (int s = 0; s < 2; ++s) {
                int kb = s * 32;
                v4f p0 = *(const v4f*)(ep + kb + hi * 8);
                v4f p1 = *(const v4f*)(ep + kb + hi * 8 + 4);
                v4f p2 = *(const v4f*)(ep + kb + 16 + hi * 8);
                v4f p3 = *(const v4f*)(ep + kb + 16 + hi * 8 + 4);
                v16bf A;
#pragma unroll
                for (int j = 0; j < 4; ++j) {
                    A[j]      = (__bf16)p0[j];
                    A[4 + j]  = (__bf16)p1[j];
                    A[8 + j]  = (__bf16)p2[j];
                    A[12 + j] = (__bf16)p3[j];
                }
                if (s == 0) A0 = A; else A1 = A;
            }
        }
#pragma unroll
        for (int t4 = 0; t4 < 4; ++t4) {
            float bv = sb0[t4 * 16 + r];
            v8f c = {bv, bv, bv, bv, bv, bv, bv, bv};
            v16bf B0 = ld_frag(&sW0T[(t4 * 16) * 64], r, 0, hi);
            v16bf B1 = ld_frag(&sW0T[(t4 * 16) * 64], r, 32, hi);
            c = wmma_bf16(A0, B0, c);
            c = wmma_bf16(A1, B1, c);
#pragma unroll
            for (int i = 0; i < 8; ++i) {
                float g = gelu_tanh(c[i]);
                sH[wave][(i + 8 * hi) * 64 + t4 * 16 + r] = (__bf16)g;
            }
        }
        asm volatile("s_wait_dscnt 0" ::: "memory");

        // A fragments of h for GEMM2 (reused for all 80 column tiles)
        v16bf HA0 = ld_frag(&sH[wave][0], r, 0, hi);
        v16bf HA1 = ld_frag(&sH[wave][0], r, 32, hi);

        // ---------------- GEMM2 + tensor-product accumulation -------------
        // q0: path0 (scale x0, *s0 at end)    q1: path1 (scale t2)
        // q2: path2 (scale x0, *s1[k] at end) q3..q5: path3 (scale x1[k], *s0 at end)
        // q6..q8: path4 rk = sum_u x1[u,k]*w4  (cross with s1 at end)
        float q[9][8];
#pragma unroll
        for (int a = 0; a < 9; ++a)
#pragma unroll
            for (int i = 0; i < 8; ++i) q[a][i] = 0.0f;

#pragma unroll 1
        for (int u = 0; u < 16; ++u) {
            float scx0[8], sct2[8], scx1[8][3];
#pragma unroll
            for (int i = 0; i < 8; ++i) {
                int e2 = i + 8 * hi;
                v2f xt  = *(const v2f*)&sXT[wave][(e2 * 16 + u) * 2];   // b64
                v4f x1v = *(const v4f*)&sX1[wave][(e2 * 16 + u) * 4];   // b128
                scx0[i] = xt[0];
                sct2[i] = xt[1];
                scx1[i][0] = x1v[0]; scx1[i][1] = x1v[1]; scx1[i][2] = x1v[2];
            }
#pragma unroll
            for (int p = 0; p < NPATH; ++p) {
                int colt = p * 16 + u;
                const __bf16* wb = W1T + (size_t)colt * 16 * 64;
                v16bf B0 = ld_frag(wb, r, 0, hi);
                v16bf B1 = ld_frag(wb, r, 32, hi);
                float bv = sb1[colt * 16 + r];
                v8f c = {bv, bv, bv, bv, bv, bv, bv, bv};
                c = wmma_bf16(HA0, B0, c);
                c = wmma_bf16(HA1, B1, c);
#pragma unroll
                for (int i = 0; i < 8; ++i) {
                    float cv = c[i];
                    if (p == 0) {
                        q[0][i] += scx0[i] * cv;
                    } else if (p == 1) {
                        q[1][i] += sct2[i] * cv;
                    } else if (p == 2) {
                        q[2][i] += scx0[i] * cv;
                    } else if (p == 3) {
                        q[3][i] += scx1[i][0] * cv;
                        q[4][i] += scx1[i][1] * cv;
                        q[5][i] += scx1[i][2] * cv;
                    } else {
                        q[6][i] += scx1[i][0] * cv;
                        q[7][i] += scx1[i][1] * cv;
                        q[8][i] += scx1[i][2] * cv;
                    }
                }
            }
        }

        // ---------------- epilogue: combine + scatter-add ------------------
#pragma unroll
        for (int i = 0; i < 8; ++i) {
            int e2 = i + 8 * hi;
            if (tb + e2 >= n_edge) continue;
            int d = sDST[wave][e2];
            v4f sv = *(const v4f*)&sSH[wave][e2 * 4];
            float s0v = sv[0], sx = sv[1], sy = sv[2], sz = sv[3];
            float o0  = A0C * (s0v * q[0][i] + q[1][i]);
            float q2v = q[2][i];
            float o10 = A1C * (sx * q2v + s0v * q[3][i] + INV_SQRT2 * (sz * q[7][i] - sy * q[8][i]));
            float o11 = A1C * (sy * q2v + s0v * q[4][i] + INV_SQRT2 * (sx * q[8][i] - sz * q[6][i]));
            float o12 = A1C * (sz * q2v + s0v * q[5][i] + INV_SQRT2 * (sy * q[6][i] - sx * q[7][i]));
            float* outp = summed + (size_t)d * 64;
            unsafeAtomicAdd(outp + r, o0);
            unsafeAtomicAdd(outp + 16 + 3 * r + 0, o10);
            unsafeAtomicAdd(outp + 16 + 3 * r + 1, o11);
            unsafeAtomicAdd(outp + 16 + 3 * r + 2, o12);
        }
    }
}

// ---------------------------------------------------------------------------
// Kernel 3: mean over edges per node, write x to d_out, reduce BN statistics
__global__ void stats_kernel(const float* __restrict__ summed,
                             const float* __restrict__ cnt,
                             float* __restrict__ outx,
                             float* __restrict__ stats,   // [48]
                             int n_dst)
{
    __shared__ float bl[48];
    int t = threadIdx.x;
    if (t < 48) bl[t] = 0.0f;
    __syncthreads();
    int n = blockIdx.x * blockDim.x + t;
    if (n < n_dst) {
        float cinv = 1.0f / fmaxf(cnt[n], 1.0f);
        const float* sp = summed + (size_t)n * 64;
        float* op = outx + (size_t)n * 64;
        for (int u = 0; u < 16; ++u) {
            float x = sp[u] * cinv;
            op[u] = x;
            atomicAdd(&bl[u], x);
            atomicAdd(&bl[16 + u], x * x);
        }
        for (int u = 0; u < 16; ++u) {
            float a = sp[16 + 3 * u + 0] * cinv;
            float b = sp[16 + 3 * u + 1] * cinv;
            float c = sp[16 + 3 * u + 2] * cinv;
            op[16 + 3 * u + 0] = a;
            op[16 + 3 * u + 1] = b;
            op[16 + 3 * u + 2] = c;
            atomicAdd(&bl[32 + u], (a * a + b * b + c * c) * (1.0f / 3.0f));
        }
    }
    __syncthreads();
    if (t < 48) unsafeAtomicAdd(&stats[t], bl[t]);
}

// Kernel 4: apply irreps batch-norm in place on d_out
__global__ void bn_kernel(float* __restrict__ outx,
                          const float* __restrict__ stats,
                          const float* __restrict__ bnw0,
                          const float* __restrict__ bnb0,
                          const float* __restrict__ bnw1,
                          int n_dst)
{
    int n = blockIdx.x * blockDim.x + threadIdx.x;
    if (n >= n_dst) return;
    float invN = 1.0f / (float)n_dst;
    float* op = outx + (size_t)n * 64;
    for (int u = 0; u < 16; ++u) {
        float mu  = stats[u] * invN;
        float var = stats[16 + u] * invN - mu * mu;
        op[u] = (op[u] - mu) * rsqrtf(var + 1e-5f) * bnw0[u] + bnb0[u];
        float nr = stats[32 + u] * invN;
        float sc = rsqrtf(nr + 1e-5f) * bnw1[u];
        op[16 + 3 * u + 0] *= sc;
        op[16 + 3 * u + 1] *= sc;
        op[16 + 3 * u + 2] *= sc;
    }
}

// ---------------------------------------------------------------------------
extern "C" void kernel_launch(void* const* d_in, const int* in_sizes, int n_in,
                              void* d_out, int out_size, void* d_ws, size_t ws_size,
                              hipStream_t stream) {
    (void)n_in; (void)ws_size;
    const float* src_features = (const float*)d_in[0];
    const float* edge_sh      = (const float*)d_in[1];
    const float* edge_scalars = (const float*)d_in[2];
    const float* W0           = (const float*)d_in[3];
    const float* b0           = (const float*)d_in[4];
    const float* W1           = (const float*)d_in[5];
    const float* b1           = (const float*)d_in[6];
    const float* bn_w0        = (const float*)d_in[7];
    const float* bn_b0        = (const float*)d_in[8];
    const float* bn_w1        = (const float*)d_in[9];
    const int*   src          = (const int*)d_in[10];
    const int*   dst          = (const int*)d_in[11];

    int n_edge = in_sizes[1] / 4;
    int n_dst  = out_size / 64;

    char* ws = (char*)d_ws;
    float* summed = (float*)ws;
    float* cntp   = summed + (size_t)n_dst * 64;
    float* stats  = cntp + n_dst;
    size_t zfloats = (size_t)n_dst * 64 + n_dst + 48;
    size_t w1t_off = ((zfloats * 4) + 255) & ~(size_t)255;
    __bf16* W1T = (__bf16*)(ws + w1t_off);

    // 0) zero accumulators (summed, cnt, stats are contiguous)
    {
        int blocks = (int)((zfloats + 511) / 512);
        if (blocks > 2048) blocks = 2048;
        zero_kernel<<<blocks, 512, 0, stream>>>(summed, zfloats);
    }
    // 1) W1 -> bf16 transposed image
    w1t_kernel<<<(64 * 1280 + 255) / 256, 256, 0, stream>>>(W1, W1T);

    // 2) fused edge kernel
    int nTiles = (n_edge + 15) / 16;
    int blocks = (nTiles + NW - 1) / NW;
    if (blocks > 8192) blocks = 8192;
    tp_edge_kernel<<<blocks, 128, 0, stream>>>(
        src_features, edge_sh, edge_scalars, W0, b0, b1, W1T,
        src, dst, summed, cntp, n_edge);

    // 3) mean + stats
    stats_kernel<<<(n_dst + 255) / 256, 256, 0, stream>>>(
        summed, cntp, (float*)d_out, stats, n_dst);

    // 4) batch-norm
    bn_kernel<<<(n_dst + 255) / 256, 256, 0, stream>>>(
        (float*)d_out, stats, bn_w0, bn_b0, bn_w1, n_dst);
}
